// RBSQuantizedLinear_64330020159887
// MI455X (gfx1250) — compile-verified, matrix operand order checked by reading, not run on hardware
//
#include <hip/hip_runtime.h>
#include <hip/hip_bf16.h>

// RBS quantized linear: out[M,P] = x[M,R] * W_deq[P,R]^T
//   M = B*S = 8192, R = 4096 (reduction, reference "N"), P = 4096 (reference "K")
// Kernel 1: dequantize int4-coded W (+ group scale/zero, sinkhorn mu1/mu2, mask) -> f16 [P,R] in d_ws (32 MB)
// Kernel 2: tiled f16 WMMA GEMM (v_wmma_f32_16x16x32_f16), f32 accumulate.
//   Block tile 128(M) x 256(P), k-tile 32, 8 waves as 2x4, wave tile 64x64 = 4x4 WMMA frags.

typedef __attribute__((ext_vector_type(16))) _Float16 v16h;
typedef __attribute__((ext_vector_type(8)))  _Float16 v8h;
typedef __attribute__((ext_vector_type(4)))  _Float16 v4h;
typedef __attribute__((ext_vector_type(8)))  float    v8f;
typedef __attribute__((ext_vector_type(4)))  float    v4f;
typedef __attribute__((ext_vector_type(4)))  int      v4i;

#define M_DIM 8192
#define R_DIM 4096   // reduction dim (reference N)
#define P_DIM 4096   // output features (reference K)
#define GROUP_SIZE 64
#define N_GROUPS (R_DIM / GROUP_SIZE)
#define TK 32
#define TM 128
#define TP 256
#define LDS_STRIDE 40   // 32 halves + 8 pad -> 80B rows, conflict-free b128 frag loads

// ---------------------------------------------------------------------------
// Kernel 1: W dequant -> f16
// One thread handles 4 consecutive r in one row p (4 | 64, so one group each).
// ---------------------------------------------------------------------------
__global__ __launch_bounds__(256) void rbs_dequant_kernel(
    const int* __restrict__ Wq, const float* __restrict__ scales,
    const float* __restrict__ zeros, const float* __restrict__ mask,
    const float* __restrict__ mu1, const float* __restrict__ mu2,
    _Float16* __restrict__ Wh) {
  const int gid = blockIdx.x * 256 + threadIdx.x;   // 0 .. P_DIM*(R_DIM/4)-1
  const int p = gid >> 10;                          // R_DIM/4 = 1024 threads per row
  const int r = (gid & 1023) * 4;
  const int g = r >> 6;                             // group index (all 4 share it)

  const float s  = scales[p * N_GROUPS + g];
  const float z  = zeros[p * N_GROUPS + g];
  const float m2 = mu2[p];

  const size_t base = (size_t)p * R_DIM + r;
  v4i q  = *(const v4i*)(Wq + base);
  v4f mk = *(const v4f*)(mask + base);
  v4f m1 = *(const v4f*)(mu1 + r);

  v4h out;
#pragma unroll
  for (int i = 0; i < 4; ++i) {
    float w = ((float)q[i] - z) * s * m2 * mk[i] * m1[i];
    out[i] = (_Float16)w;
  }
  *(v4h*)(Wh + base) = out;
}

// ---------------------------------------------------------------------------
// Kernel 2: GEMM  out[M,P] = X[M,R] * Wh[P,R]^T  via v_wmma_f32_16x16x32_f16
// 128x256 tile per 256-thread block, k-tile 32, double-buffered LDS.
// Wave (8 total): wm in {0,1} -> 64-row strip, wp in {0..3} -> 64-col strip;
// each wave holds 4x4 16x16 f32 accumulators (16 WMMAs per k-step,
// 16 ds_load_b128 per k-step -> 1:1 LDS:XDL issue ratio).
// ---------------------------------------------------------------------------
__global__ __launch_bounds__(256) void rbs_wmma_gemm_kernel(
    const float* __restrict__ X, const _Float16* __restrict__ Wh,
    float* __restrict__ out) {
  __shared__ _Float16 Xs[2][TM * LDS_STRIDE];   // 10 KB each
  __shared__ _Float16 Ws[2][TP * LDS_STRIDE];   // 20 KB each; total 60 KB

  const int tid  = threadIdx.x;
  const int lane = tid & 31;
  const int wid  = tid >> 5;
  const int half = lane >> 4;   // 0: K 0..7 & 16..23 ; 1: K 8..15 & 24..31
  const int lrow = lane & 15;
  const int wm   = wid >> 2;    // 0..1  -> 64-row strip of the 128-row tile
  const int wp   = wid & 3;     // 0..3  -> 64-col strip of the 256-col tile

  const int m0 = blockIdx.y * TM;
  const int p0 = blockIdx.x * TP;

  // staging assignments (global -> regs -> LDS)
  const int xRow = tid >> 3;        // 0..31, +32*i covers 128 rows
  const int xCol = (tid & 7) * 4;   // f32 cols 0..28 step 4
  const int wRow = tid >> 2;        // 0..63, +64*i covers 256 rows
  const int wCol = (tid & 3) * 8;   // f16 cols 0..24 step 8

  const float*    xg = X  + (size_t)(m0 + xRow) * R_DIM + xCol;
  const _Float16* wg = Wh + (size_t)(p0 + wRow) * R_DIM + wCol;

  v4f xr[4];
  v8h wr[4];

  auto loadG = [&](int k0) {
#pragma unroll
    for (int i = 0; i < 4; ++i)
      xr[i] = *(const v4f*)(xg + (size_t)(i * 32) * R_DIM + k0);
#pragma unroll
    for (int i = 0; i < 4; ++i)
      wr[i] = *(const v8h*)(wg + (size_t)(i * 64) * R_DIM + k0);
  };
  auto storeL = [&](int buf) {
#pragma unroll
    for (int i = 0; i < 4; ++i) {
      v4h h;
#pragma unroll
      for (int e = 0; e < 4; ++e) h[e] = (_Float16)xr[i][e];
      *(v4h*)(&Xs[buf][(xRow + i * 32) * LDS_STRIDE + xCol]) = h;
    }
#pragma unroll
    for (int i = 0; i < 4; ++i)
      *(v8h*)(&Ws[buf][(wRow + i * 64) * LDS_STRIDE + wCol]) = wr[i];
  };

  // loop-invariant fragment offsets (in halves). Per-lane layout per ISA:
  // chunk0 at row*stride + half*8, chunk1 at row*stride + 16 + half*8.
  int aOff[4], bOff[4];
#pragma unroll
  for (int i = 0; i < 4; ++i)
    aOff[i] = (wm * 64 + i * 16 + lrow) * LDS_STRIDE + half * 8;
#pragma unroll
  for (int j = 0; j < 4; ++j)
    bOff[j] = (wp * 64 + j * 16 + lrow) * LDS_STRIDE + half * 8;

  v8f acc[4][4];
  const v8f zero = {};
#pragma unroll
  for (int i = 0; i < 4; ++i)
#pragma unroll
    for (int j = 0; j < 4; ++j) acc[i][j] = zero;

  loadG(0);
  storeL(0);
  __syncthreads();

  const int KT = R_DIM / TK;   // 128
  for (int kt = 0; kt < KT; ++kt) {
    const int buf = kt & 1;
    if (kt + 1 < KT) loadG((kt + 1) * TK);

    v16h a[4], b[4];
#pragma unroll
    for (int i = 0; i < 4; ++i) {
      v8h lo = *(const v8h*)(&Xs[buf][aOff[i]]);
      v8h hi = *(const v8h*)(&Xs[buf][aOff[i] + 16]);
      a[i] = __builtin_shufflevector(lo, hi, 0, 1, 2, 3, 4, 5, 6, 7,
                                             8, 9, 10, 11, 12, 13, 14, 15);
    }
#pragma unroll
    for (int j = 0; j < 4; ++j) {
      v8h lo = *(const v8h*)(&Ws[buf][bOff[j]]);
      v8h hi = *(const v8h*)(&Ws[buf][bOff[j] + 16]);
      b[j] = __builtin_shufflevector(lo, hi, 0, 1, 2, 3, 4, 5, 6, 7,
                                             8, 9, 10, 11, 12, 13, 14, 15);
    }

#pragma unroll
    for (int i = 0; i < 4; ++i)
#pragma unroll
      for (int j = 0; j < 4; ++j)
        acc[i][j] = __builtin_amdgcn_wmma_f32_16x16x32_f16(
            false, a[i], false, b[j], (short)0, acc[i][j], false, false);

    if (kt + 1 < KT) {
      storeL(buf ^ 1);
      __syncthreads();
    }
  }

  // Epilogue: C/D layout -> lane%16 = column, VGPR v = row v (+8 for lanes>=16)
#pragma unroll
  for (int i = 0; i < 4; ++i) {
#pragma unroll
    for (int j = 0; j < 4; ++j) {
      const int col  = p0 + wp * 64 + j * 16 + lrow;
      const int rowB = m0 + wm * 64 + i * 16 + half * 8;
#pragma unroll
      for (int v = 0; v < 8; ++v)
        out[(size_t)(rowB + v) * P_DIM + col] = acc[i][j][v];
    }
  }
}

extern "C" void kernel_launch(void* const* d_in, const int* in_sizes, int n_in,
                              void* d_out, int out_size, void* d_ws, size_t ws_size,
                              hipStream_t stream) {
  const float* x      = (const float*)d_in[0];
  const int*   Wq     = (const int*)d_in[1];
  const float* scales = (const float*)d_in[2];
  const float* zeros  = (const float*)d_in[3];
  const float* mask   = (const float*)d_in[4];
  const float* mu1    = (const float*)d_in[5];
  const float* mu2    = (const float*)d_in[6];
  float* out = (float*)d_out;
  _Float16* Wh = (_Float16*)d_ws;   // needs P_DIM*R_DIM*2 = 32 MB of workspace

  // Kernel 1: dequant W -> f16
  const int dq_threads = P_DIM * (R_DIM / 4);       // 4,194,304
  rbs_dequant_kernel<<<dq_threads / 256, 256, 0, stream>>>(
      Wq, scales, zeros, mask, mu1, mu2, Wh);

  // Kernel 2: WMMA GEMM
  dim3 grid(P_DIM / TP, M_DIM / TM);                // 16 x 64
  rbs_wmma_gemm_kernel<<<grid, 256, 0, stream>>>(x, Wh, out);
}
